// VQCLayer_48576080118730
// MI455X (gfx1250) — matrix operand order repaired and checked by reading, not compiled, not run on hardware
//
#include <hip/hip_runtime.h>
#include <math.h>

typedef __attribute__((ext_vector_type(16))) _Float16 v16h;
typedef __attribute__((ext_vector_type(8)))  float    v8f;

#define NQ   12
#define NL   4
#define QDIM 4096   // 2^12

#define WMMA_CF16(A, B) \
    __builtin_amdgcn_wmma_f32_16x16x32_f16(false, (A), false, (B), (short)0, zro, false, false)

// ---------------------------------------------------------------------------
// CNOT ring permutation. new[m] = old[F(m)], F = f_0 o f_1 o ... o f_11,
// f_q: flip target bit if control bit set.  F is GF(2)-linear.
// qubit q <-> index bit (11-q); CNOT(q, (q+1)%12).
// ---------------------------------------------------------------------------
__device__ __forceinline__ int permF(int v) {
#pragma unroll
    for (int q = 11; q >= 0; --q) {
        const int c = 11 - q;
        const int t = (q == 11) ? 11 : (10 - q);
        v ^= ((v >> c) & 1) << t;
    }
    return v;
}
__device__ __forceinline__ int permFinv(int v) {
#pragma unroll
    for (int q = 0; q <= 11; ++q) {
        const int c = 11 - q;
        const int t = (q == 11) ? 11 : (10 - q);
        v ^= ((v >> c) & 1) << t;
    }
    return v;
}

// ---------------------------------------------------------------------------
// Build the 12 fused 4-qubit RX unitaries (4 layers x 3 groups), 16x16 complex.
// U[j,i] = (-i)^popcount(j^i) * prod_m ( bit_m(j)==bit_m(i) ? cos : sin )(w/2)
// d_ws layout: [l*3+g][ Ure[16][16] , Uim[16][16] ] (512 floats each)
// ---------------------------------------------------------------------------
__global__ void vqc_build_gates(const float* __restrict__ w,
                                float* __restrict__ mats) {
    const int lg = blockIdx.x;
    const int l = lg / 3, g = lg % 3;
    const int t = threadIdx.x;          // j*16+i
    const int j = t >> 4, i = t & 15;
    float mag = 1.0f;
#pragma unroll
    for (int m = 0; m < 4; ++m) {
        const int q = 4 * g + (3 - m);  // sub-bit m <-> qubit 4g+(3-m)
        const float th = 0.5f * w[l * NQ + q];
        const float c = cosf(th), s = sinf(th);
        mag *= (((j >> m) & 1) == ((i >> m) & 1)) ? c : s;
    }
    const int h = __popc(j ^ i) & 3;
    float re = 0.0f, im = 0.0f;
    if      (h == 0) re =  mag;
    else if (h == 1) im = -mag;
    else if (h == 2) re = -mag;
    else             im =  mag;
    float* base = mats + lg * 512;
    base[t]       = re;
    base[256 + t] = im;
}

// ======================= fused gate-group macros ===========================
// Complex 16x16 gate x state packed into one K=32 f16 WMMA pair:
//   out_re = [G_re | -G_im] x [S_re ; S_im],  out_im = [G_im | G_re] x [S_re ; S_im]

// group 0: qubits 0-3 (index bits 11-8); gate = A operand, state = B operand.
// Tiles partition cols r (bits 7-0).  Optional gather applies deferred CNOT perm.
#define GROUP0(L, SR, SI, DR, DI, GATHER)                                      \
    {                                                                          \
        const float* Ure_ = mats + ((L) * 3 + 0) * 512;                        \
        const float* Uim_ = Ure_ + 256;                                        \
        v16h Are, Aim;                                                         \
        _Pragma("unroll")                                                      \
        for (int e = 0; e < 16; ++e) {                                         \
            const float ur = Ure_[lo * 16 + e];                                \
            const float ui = Uim_[lo * 16 + e];                                \
            Are[e] = (_Float16)(hi ? -ui : ur);                                \
            Aim[e] = (_Float16)(hi ? ur : ui);                                 \
        }                                                                      \
        _Pragma("unroll")                                                      \
        for (int it = 0; it < 2; ++it) {                                       \
            const int r0 = (wav + 8 * it) * 16;                                \
            const int bidx = (GATHER) ? (permF(r0) ^ Flo) : (r0 + lo);         \
            const float* sp_ = hi ? (SI) : (SR);                               \
            v16h B;                                                            \
            _Pragma("unroll")                                                  \
            for (int e = 0; e < 16; ++e) {                                     \
                const int idx = bidx ^ ((GATHER) ? Fhigh[e] : (e << 8));       \
                B[e] = (_Float16)sp_[idx];                                     \
            }                                                                  \
            v8f dre = WMMA_CF16(Are, B);                                       \
            v8f dIm = WMMA_CF16(Aim, B);                                       \
            _Pragma("unroll")                                                  \
            for (int v = 0; v < 8; ++v) {                                      \
                const int k = ((v + 8 * hi) << 8) | (r0 + lo);                 \
                (DR)[k] = dre[v];                                              \
                (DI)[k] = dIm[v];                                              \
            }                                                                  \
        }                                                                      \
        __syncthreads();                                                       \
    }

// group 1: qubits 4-7 (bits 7-4); gate = A, state = B. Tiles partition bits 11-8.
#define GROUP1(L, SR, SI, DR, DI)                                              \
    {                                                                          \
        const float* Ure_ = mats + ((L) * 3 + 1) * 512;                        \
        const float* Uim_ = Ure_ + 256;                                        \
        v16h Are, Aim;                                                         \
        _Pragma("unroll")                                                      \
        for (int e = 0; e < 16; ++e) {                                         \
            const float ur = Ure_[lo * 16 + e];                                \
            const float ui = Uim_[lo * 16 + e];                                \
            Are[e] = (_Float16)(hi ? -ui : ur);                                \
            Aim[e] = (_Float16)(hi ? ur : ui);                                 \
        }                                                                      \
        _Pragma("unroll")                                                      \
        for (int it = 0; it < 2; ++it) {                                       \
            const int u = (wav + 8 * it) << 8;                                 \
            const float* sp_ = hi ? (SI) : (SR);                               \
            v16h B;                                                            \
            _Pragma("unroll")                                                  \
            for (int e = 0; e < 16; ++e)                                       \
                B[e] = (_Float16)sp_[u | (e << 4) | lo];                       \
            v8f dre = WMMA_CF16(Are, B);                                       \
            v8f dIm = WMMA_CF16(Aim, B);                                       \
            _Pragma("unroll")                                                  \
            for (int v = 0; v < 8; ++v) {                                      \
                const int k = u | ((v + 8 * hi) << 4) | lo;                    \
                (DR)[k] = dre[v];                                              \
                (DI)[k] = dIm[v];                                              \
            }                                                                  \
        }                                                                      \
        __syncthreads();                                                       \
    }

// group 2: qubits 8-11 (bits 3-0); state = A, gate = B. Tiles partition bits 11-4.
// LAST: fold final deferred permutation + <Z_q> reduction; no LDS writeback.
#define GROUP2(L, SR, SI, DR, DI, LAST)                                        \
    {                                                                          \
        const float* Ure_ = mats + ((L) * 3 + 2) * 512;                        \
        const float* Uim_ = Ure_ + 256;                                        \
        v16h Bre, Bim;  /* B[k,j]: k<16 -> Ure[j,k] ; k>=16 -> -Uim[j,k-16] */ \
        _Pragma("unroll")                                                      \
        for (int e = 0; e < 16; ++e) {                                         \
            const float ur = Ure_[lo * 16 + e];                                \
            const float ui = Uim_[lo * 16 + e];                                \
            Bre[e] = (_Float16)(hi ? -ui : ur);                                \
            Bim[e] = (_Float16)(hi ? ur : ui);                                 \
        }                                                                      \
        _Pragma("unroll")                                                      \
        for (int it = 0; it < 2; ++it) {                                       \
            const int tt = wav + 8 * it;                                       \
            const int obase = (tt * 16 + lo) * 16;                             \
            v16h A;                                                            \
            _Pragma("unroll")                                                  \
            for (int e = 0; e < 16; ++e) {                                     \
                A[e]     = (_Float16)(hi ? (SI) : (SR))[obase + e];            \
            }                                                                  \
            v8f dre = WMMA_CF16(A, Bre);                                       \
            v8f dIm = WMMA_CF16(A, Bim);                                       \
            if (LAST) {                                                        \
                _Pragma("unroll")                                              \
                for (int v = 0; v < 8; ++v) {                                  \
                    const int k = ((tt * 16 + v + 8 * hi) << 4) | lo;          \
                    const int m = permFinv(k); /* final perm position */       \
                    const float p = dre[v] * dre[v] + dIm[v] * dIm[v];         \
                    _Pragma("unroll")                                          \
                    for (int q = 0; q < NQ; ++q)                               \
                        part[q] += ((m >> (11 - q)) & 1) ? -p : p;             \
                }                                                              \
            } else {                                                           \
                _Pragma("unroll")                                              \
                for (int v = 0; v < 8; ++v) {                                  \
                    const int k = ((tt * 16 + v + 8 * hi) << 4) | lo;          \
                    (DR)[k] = dre[v];                                          \
                    (DI)[k] = dIm[v];                                          \
                }                                                              \
            }                                                                  \
        }                                                                      \
        if (!(LAST)) __syncthreads();                                          \
    }

// ---------------------------------------------------------------------------
// One workgroup (256 threads = 8 wave32) simulates one batch element.
// Ping-pong f32 state in LDS; deferred CNOT perms folded into gathers (F linear).
// ---------------------------------------------------------------------------
__global__ __launch_bounds__(256) void vqc_sim(const float* __restrict__ x,
                                               const float* __restrict__ mats,
                                               float* __restrict__ out) {
    __shared__ float R0[QDIM + 16];   // +16 staggers bases across bank groups
    __shared__ float I0[QDIM + 16];
    __shared__ float R1[QDIM + 16];
    __shared__ float I1[QDIM + 16];
    __shared__ float cs[NQ][2];
    __shared__ float zacc[NQ];

    const int b    = blockIdx.x;
    const int tid  = threadIdx.x;
    const int lane = tid & 31;
    const int wav  = tid >> 5;
    const int hi   = lane >> 4;
    const int lo   = lane & 15;

    if (tid < NQ) {
        const float th = 0.5f * x[b * NQ + tid];
        cs[tid][0] = cosf(th);
        cs[tid][1] = sinf(th);
        zacc[tid]  = 0.0f;
    }
    __syncthreads();

    // ---- RY layer on |0..0> == closed-form product state (real) ----
    float baseP = 1.0f;
#pragma unroll
    for (int q = 4; q < NQ; ++q) baseP *= cs[q][(tid >> (11 - q)) & 1];
#pragma unroll
    for (int e = 0; e < 16; ++e) {
        const float topv = cs[0][(e >> 3) & 1] * cs[1][(e >> 2) & 1] *
                           cs[2][(e >> 1) & 1] * cs[3][e & 1];
        const int k = (e << 8) | tid;
        R0[k] = topv * baseP;
        I0[k] = 0.0f;
    }

    // ---- permutation basis images (F is linear over GF(2)) ----
    int Fhigh[16];
#pragma unroll
    for (int e = 0; e < 16; ++e) Fhigh[e] = permF(e << 8);
    const int Flo = permF(lo);
    __syncthreads();

    const v8f zro = {};
    float part[NQ];
#pragma unroll
    for (int q = 0; q < NQ; ++q) part[q] = 0.0f;

    // ---- 4 layers, straight-line, ping-pong buffers.
    //      G0 of layers 1..3 applies the previous layer's deferred CNOT perm
    //      via gathered reads; layer 3's G2 reduces directly from registers. ----
    GROUP0(0, R0, I0, R1, I1, false)
    GROUP1(0, R1, I1, R0, I0)
    GROUP2(0, R0, I0, R1, I1, false)

    GROUP0(1, R1, I1, R0, I0, true)
    GROUP1(1, R0, I0, R1, I1)
    GROUP2(1, R1, I1, R0, I0, false)

    GROUP0(2, R0, I0, R1, I1, true)
    GROUP1(2, R1, I1, R0, I0)
    GROUP2(2, R0, I0, R1, I1, false)

    GROUP0(3, R1, I1, R0, I0, true)
    GROUP1(3, R0, I0, R1, I1)
    GROUP2(3, R1, I1, R0, I0, true)   // LAST: reduce, no store

    // ---- wave32 shuffle reduce + LDS atomics ----
#pragma unroll
    for (int off = 16; off > 0; off >>= 1) {
#pragma unroll
        for (int q = 0; q < NQ; ++q)
            part[q] += __shfl_down(part[q], off, 32);
    }
    if (lane == 0) {
#pragma unroll
        for (int q = 0; q < NQ; ++q) atomicAdd(&zacc[q], part[q]);
    }
    __syncthreads();
    if (tid < NQ) out[b * NQ + tid] = zacc[tid];
}

// ---------------------------------------------------------------------------
extern "C" void kernel_launch(void* const* d_in, const int* in_sizes, int n_in,
                              void* d_out, int out_size, void* d_ws, size_t ws_size,
                              hipStream_t stream) {
    const float* x = (const float*)d_in[0];   // (B, 12) f32
    const float* w = (const float*)d_in[1];   // (4, 12) f32
    float* out  = (float*)d_out;              // (B, 12) f32
    float* mats = (float*)d_ws;               // 12 * 512 floats = 24 KB

    const int batch = in_sizes[0] / NQ;

    vqc_build_gates<<<dim3(NL * 3), dim3(256), 0, stream>>>(w, mats);
    vqc_sim<<<dim3(batch), dim3(256), 0, stream>>>(x, mats, out);
}
